// TransformerAttention_23759759081760
// MI455X (gfx1250) — matrix operand order repaired and checked by reading, not compiled, not run on hardware
//
#include <hip/hip_runtime.h>

typedef __attribute__((ext_vector_type(16))) _Float16 v16h;
typedef __attribute__((ext_vector_type(8)))  _Float16 v8h;
typedef __attribute__((ext_vector_type(4)))  _Float16 v4h;
typedef __attribute__((ext_vector_type(2)))  _Float16 v2h;
typedef __attribute__((ext_vector_type(8)))  float    v8f;
typedef __attribute__((ext_vector_type(4)))  float    v4f;

#define B_     8
#define H_     16
#define L_     1024
#define D_     64
#define QTILE  64     // query rows per workgroup (16 per wave)
#define KCHUNK 32     // keys per iteration

#if __has_builtin(__builtin_amdgcn_s_wait_asynccnt)
#define WAIT_ASYNC(n) __builtin_amdgcn_s_wait_asynccnt(n)
#else
#define WAIT_ASYNC(n) asm volatile("s_wait_asynccnt %0" :: "i"(n) : "memory")
#endif

// ---------------- pre-pass: K -> f16 (same layout), V -> f16 transposed ----------------
// kh [bh][key][d], vth [bh][d][key]; 64-key tiles transposed through LDS.
__global__ __launch_bounds__(256) void cvt_kv_f16(
    const float* __restrict__ k, const float* __restrict__ v,
    _Float16* __restrict__ kh, _Float16* __restrict__ vth)
{
    __shared__ _Float16 tile[D_ * 64];            // [d][key] for one 64-key tile
    const int tid = threadIdx.x;
    const int bh  = blockIdx.x >> 4;              // 16 tiles per (b,h)
    const int kt  = blockIdx.x & 15;

    const float* kb  = k  + ((size_t)bh * L_ + kt * 64) * D_;
    const float* vb  = v  + ((size_t)bh * L_ + kt * 64) * D_;
    _Float16*    khb = kh + ((size_t)bh * L_ + kt * 64) * D_;

    const int key = tid >> 2, d0 = (tid & 3) * 16;   // 16 contiguous floats per thread
    #pragma unroll
    for (int i = 0; i < 16; i += 4) {
        v4f kf = *(const v4f*)(kb + key * D_ + d0 + i);
        v4f vf = *(const v4f*)(vb + key * D_ + d0 + i);
        v4h kp;
        #pragma unroll
        for (int j = 0; j < 4; ++j) {
            kp[j] = (_Float16)kf[j];
            tile[(d0 + i + j) * 64 + key] = (_Float16)vf[j];   // transpose in LDS
        }
        *(v4h*)(khb + key * D_ + d0 + i) = kp;
    }
    __syncthreads();
    const int d = tid >> 2, s0 = (tid & 3) * 16;     // write V^T rows coalesced
    _Float16* vo = vth + (size_t)bh * L_ * D_ + (size_t)d * L_ + kt * 64 + s0;
    *(v8h*)(vo)     = *(const v8h*)&tile[d * 64 + s0];
    *(v8h*)(vo + 8) = *(const v8h*)&tile[d * 64 + s0 + 8];
}

// ---------------- main kernel: transposed flash attention ----------------
//   S^T = K * Q^T   (D-matrix: keys x m, m on lane axis)
//   O^T = V^T * P^T (D-matrix: d x m,    m on lane axis)
// K/V chunks staged LDS-side via async-to-LDS, double buffered.
__global__ __launch_bounds__(128) void attn_flash_wmma_async_gfx1250(
    const float* __restrict__ q, const unsigned char* __restrict__ mask,
    const float* __restrict__ bias, const _Float16* __restrict__ kh,
    const _Float16* __restrict__ vth, float* __restrict__ out)
{
    __shared__ _Float16 ldsK[2][KCHUNK * D_];   // [key][d]
    __shared__ _Float16 ldsV[2][D_ * KCHUNK];   // [d][key]

    const int tid  = threadIdx.x;
    const int wave = tid >> 5;
    const int lane = tid & 31;
    const int hs   = lane >> 4;
    const int ln16 = lane & 15;

    const int bh    = blockIdx.x >> 4;
    const int qTile = blockIdx.x & 15;
    const int b     = bh >> 4;
    const int h     = bh & 15;
    const int qBase = qTile * QTILE + wave * 16;
    const int m     = qBase + ln16;             // lane fully owns this query row

    // async staging: 4 x b128 per thread (2 for K rows, 2 for a V^T row segment)
    const _Float16* gkT = kh  + (size_t)bh * (L_ * D_) + tid * 16;
    const _Float16* gvT = vth + (size_t)bh * (L_ * D_) + (size_t)(tid >> 1) * L_ + (tid & 1) * 16;
    const unsigned  lK0 = (unsigned)(uintptr_t)&ldsK[0][tid * 16];
    const unsigned  lV0 = (unsigned)(uintptr_t)&ldsV[0][(tid >> 1) * KCHUNK + (tid & 1) * 16];
    const unsigned  lK1 = (unsigned)(uintptr_t)&ldsK[1][tid * 16];
    const unsigned  lV1 = (unsigned)(uintptr_t)&ldsV[1][(tid >> 1) * KCHUNK + (tid & 1) * 16];

    auto stage = [&](int kcN, int bufSel) {
        unsigned long long ga = (unsigned long long)(uintptr_t)(gkT + (size_t)kcN * D_);
        unsigned long long gv = (unsigned long long)(uintptr_t)(gvT + kcN);
        unsigned lk = bufSel ? lK1 : lK0;
        unsigned lv = bufSel ? lV1 : lV0;
        asm volatile(
            "global_load_async_to_lds_b128 %0, %1, off\n\t"
            "global_load_async_to_lds_b128 %0, %1, off offset:16\n\t"
            "global_load_async_to_lds_b128 %2, %3, off\n\t"
            "global_load_async_to_lds_b128 %2, %3, off offset:16"
            :: "v"(lk), "v"(ga), "v"(lv), "v"(gv) : "memory");
    };

    // ---- Q^T as WMMA B operand (K-dim = d, N = m) ----
    const float* qrow = q + ((size_t)bh * L_ + m) * D_;
    v16h bq0, bq1;
    #pragma unroll
    for (int e = 0; e < 16; e += 4) {
        v4f f0 = *(const v4f*)(qrow + hs * 8 + e + (e & 8));
        v4f f1 = *(const v4f*)(qrow + 32 + hs * 8 + e + (e & 8));
        #pragma unroll
        for (int j = 0; j < 4; ++j) {
            bq0[e + j] = (_Float16)f0[j];
            bq1[e + j] = (_Float16)f1[j];
        }
    }

    float mrow = -3.0e38f, lrow = 0.0f;         // scalar online-softmax state
    v8f accO[4];
    #pragma unroll
    for (int t = 0; t < 4; ++t) accO[t] = (v8f){0,0,0,0,0,0,0,0};

    stage(0, 0);
    int buf = 0;

    for (int kc = 0; kc < L_; kc += KCHUNK) {
        __syncthreads();                         // everyone done reading buf^1
        const bool more = (kc + KCHUNK) < L_;
        if (more) { stage(kc + KCHUNK, buf ^ 1); WAIT_ASYNC(4); }
        else      { WAIT_ASYNC(0); }
        __syncthreads();                         // chunk kc visible to all waves
        const _Float16* lk = &ldsK[buf][0];
        const _Float16* lv = &ldsV[buf][0];

        // mask words for this half's keys: st[0] keys kc+hs*8+0..7, st[1] +16
        const uint2 mw0 = *(const uint2*)(mask + b * L_ + kc + hs * 8);
        const uint2 mw1 = *(const uint2*)(mask + b * L_ + kc + 16 + hs * 8);

        // ---- S^T = K*Q^T + bias; two (16key x 16m) subtiles ----
        v8f st[2];
        #pragma unroll
        for (int sub = 0; sub < 2; ++sub) {
            const float* bptr = bias + ((size_t)h * L_ + m) * L_ + kc + sub * 16 + hs * 8;
            v4f c0 = *(const v4f*)bptr;
            v4f c1 = *(const v4f*)(bptr + 4);
            v8f c;
            #pragma unroll
            for (int j = 0; j < 4; ++j) { c[j] = c0[j]; c[4 + j] = c1[j]; }

            const _Float16* krow = lk + (sub * 16 + ln16) * D_;
            const int kb0 = hs * 8;
            v16h ak0, ak1;
            #pragma unroll
            for (int e = 0; e < 8; ++e) {
                ak0[e]     = krow[kb0 + e];
                ak0[8 + e] = krow[kb0 + 16 + e];
                ak1[e]     = krow[32 + kb0 + e];
                ak1[8 + e] = krow[32 + kb0 + 16 + e];
            }
            c = __builtin_amdgcn_wmma_f32_16x16x32_f16(false, ak0, false, bq0,
                                                       (short)0, c, false, false);
            c = __builtin_amdgcn_wmma_f32_16x16x32_f16(false, ak1, false, bq1,
                                                       (short)0, c, false, false);
            st[sub] = c;
        }

        // ---- online softmax (mask folded in as 0/1 multiplier after exp) ----
        float mx = fmaxf(st[0][0], st[1][0]);
        #pragma unroll
        for (int r = 1; r < 8; ++r) mx = fmaxf(mx, fmaxf(st[0][r], st[1][r]));
        mx = fmaxf(mx, __shfl_xor(mx, 16, 32));
        const float mnew = fmaxf(mrow, mx);
        const float corr = __expf(mrow - mnew);
        float ps = 0.0f;
        #pragma unroll
        for (int r = 0; r < 8; ++r) {
            const unsigned w0 = (r < 4) ? mw0.x : mw0.y;
            const unsigned w1 = (r < 4) ? mw1.x : mw1.y;
            const float f0 = (float)((w0 >> (8 * (r & 3))) & 0xffu);  // v_cvt_f32_ubyte
            const float f1 = (float)((w1 >> (8 * (r & 3))) & 0xffu);
            const float p0 = __expf(st[0][r] - mnew) * f0;
            const float p1 = __expf(st[1][r] - mnew) * f1;
            st[0][r] = p0; st[1][r] = p1;
            ps += p0 + p1;
        }
        ps += __shfl_xor(ps, 16, 32);
        lrow = lrow * corr + ps;
        mrow = mnew;
        #pragma unroll
        for (int t = 0; t < 4; ++t)
            #pragma unroll
            for (int r = 0; r < 8; ++r) accO[t][r] *= corr;

        // ---- P^T as WMMA B operand: half-swap via 4 packed shuffles ----
        int ownw[4], sendw[4], recvw[4];
        #pragma unroll
        for (int i = 0; i < 4; ++i) {
            v2h po, pse;
            po[0]  = (_Float16)(hs ? st[1][2 * i]     : st[0][2 * i]);
            po[1]  = (_Float16)(hs ? st[1][2 * i + 1] : st[0][2 * i + 1]);
            pse[0] = (_Float16)(hs ? st[0][2 * i]     : st[1][2 * i]);
            pse[1] = (_Float16)(hs ? st[0][2 * i + 1] : st[1][2 * i + 1]);
            __builtin_memcpy(&ownw[i], &po, 4);
            __builtin_memcpy(&sendw[i], &pse, 4);
        }
        #pragma unroll
        for (int i = 0; i < 4; ++i) recvw[i] = __shfl_xor(sendw[i], 16, 32);
        union { v16h hv; int w[8]; } bp;
        #pragma unroll
        for (int i = 0; i < 4; ++i) {
            bp.w[i]     = hs ? recvw[i] : ownw[i];
            bp.w[4 + i] = hs ? ownw[i]  : recvw[i];
        }

        // ---- O^T += V^T * P^T ----
        #pragma unroll
        for (int t = 0; t < 4; ++t) {
            const _Float16* vrow = lv + (t * 16 + ln16) * KCHUNK;
            const int kb0 = hs * 16;
            v16h av;
            #pragma unroll
            for (int e = 0; e < 8; ++e) {
                av[e]     = vrow[kb0 + e];
                av[8 + e] = vrow[kb0 + 8 + e];
            }
            accO[t] = __builtin_amdgcn_wmma_f32_16x16x32_f16(false, av, false, bp.hv,
                                                             (short)0, accO[t], false, false);
        }
        buf ^= 1;
    }

    // ---- normalize, query-row mask, store [b, m, h*64 + d] ----
    float inv = (lrow > 0.0f) ? (1.0f / lrow) : 0.0f;
    if (!mask[b * L_ + m]) inv = 0.0f;
    float* orow = out + ((size_t)(b * L_) + m) * (H_ * D_) + h * D_;
    #pragma unroll
    for (int t = 0; t < 4; ++t) {
        v4f o0, o1;
        #pragma unroll
        for (int j = 0; j < 4; ++j) { o0[j] = accO[t][j] * inv; o1[j] = accO[t][4 + j] * inv; }
        *(v4f*)(orow + t * 16 + hs * 8)     = o0;
        *(v4f*)(orow + t * 16 + hs * 8 + 4) = o1;
    }
}

extern "C" void kernel_launch(void* const* d_in, const int* in_sizes, int n_in,
                              void* d_out, int out_size, void* d_ws, size_t ws_size,
                              hipStream_t stream) {
    const float*         q    = (const float*)d_in[0];
    const float*         k    = (const float*)d_in[1];
    const float*         v    = (const float*)d_in[2];
    const unsigned char* mask = (const unsigned char*)d_in[3];
    const float*         bias = (const float*)d_in[4];
    float*               out  = (float*)d_out;

    const size_t nKV = (size_t)B_ * H_ * L_ * D_;          // 8.4M elems
    _Float16* kh  = (_Float16*)d_ws;                        // 16 MB
    _Float16* vth = (_Float16*)d_ws + nKV;                  // 16 MB

    cvt_kv_f16<<<dim3(B_ * H_ * (L_ / 64)), 256, 0, stream>>>(k, v, kh, vth);
    attn_flash_wmma_async_gfx1250<<<dim3(B_ * H_ * (L_ / QTILE)), 128, 0, stream>>>(
        q, mask, bias, kh, vth, out);
}
